// SSM_88270167867995
// MI455X (gfx1250) — matrix operand order repaired and checked by reading, not compiled, not run on hardware
//
#include <hip/hip_runtime.h>

// ---------------------------------------------------------------------------
// Types
// ---------------------------------------------------------------------------
typedef __attribute__((ext_vector_type(4)))  _Float16 v4h;
typedef __attribute__((ext_vector_type(8)))  _Float16 v8h;
typedef __attribute__((ext_vector_type(16))) _Float16 v16h;
typedef __attribute__((ext_vector_type(4)))  float    v4f;
typedef __attribute__((ext_vector_type(8)))  float    v8f;

#define DIN   128
#define DOUT  128
#define DST   256
#define DSC   512
#define BSZ   131072
#define MTILE 64
#define CHUNK 256   // DSC processed in two 256-wide chunks

// ---------------------------------------------------------------------------
// Helpers
// ---------------------------------------------------------------------------
__device__ __forceinline__ v16h cat16(v8h lo, v8h hi) {
  return __builtin_shufflevector(lo, hi, 0,1,2,3,4,5,6,7,8,9,10,11,12,13,14,15);
}

// A-matrix fragment (16x32 f16): lane holds row m; VGPR0..3 = K[kb..kb+8),
// VGPR4..7 = K[16+kb..16+kb+8), kb = (lane<16)?0:8.  base = row ptr at k0.
__device__ __forceinline__ v16h load_a(const _Float16* base, int kb) {
  v8h lo = *(const v8h*)(base + kb);
  v8h hi = *(const v8h*)(base + 16 + kb);
  return cat16(lo, hi);
}

// B-matrix fragment (32x16 f16), weights stored row-major [N][K] (== B^T):
// lane holds column n; same K order as load_a: K[kb..kb+8), K[16+kb..16+kb+8),
// kb = nb/2.  base = weight-row ptr at k0, nb = (lane<16)?0:16.
__device__ __forceinline__ v16h load_b(const _Float16* base, int nb) {
  v8h lo = *(const v8h*)(base + (nb >> 1));
  v8h hi = *(const v8h*)(base + 16 + (nb >> 1));
  return cat16(lo, hi);
}

__device__ __forceinline__ v8f wmma_f16(v16h a, v16h b, v8f c) {
  return __builtin_amdgcn_wmma_f32_16x16x32_f16(false, a, false, b,
                                                (short)0, c, false, false);
}

// ---------------------------------------------------------------------------
// Precompute 1: A = D + 2*C_r*diag(gamma)*B_r - 2*C_i*diag(gamma)*B_i   [128x128]
// ---------------------------------------------------------------------------
__global__ void SSM_precompA(const float* __restrict__ D,
                             const float* __restrict__ gl,
                             const float* __restrict__ Br,
                             const float* __restrict__ Bi,
                             const float* __restrict__ Cr,
                             const float* __restrict__ Ci,
                             float* __restrict__ A) {
  int idx = blockIdx.x * blockDim.x + threadIdx.x;   // 16384
  int o = idx >> 7, i = idx & 127;
  float acc = D[idx];
  for (int s = 0; s < DST; ++s) {
    float g = expf(gl[s]);
    acc += 2.0f * g * (Cr[o * DST + s] * Br[s * DIN + i] -
                       Ci[o * DST + s] * Bi[s * DIN + i]);
  }
  A[idx] = acc;
}

// ---------------------------------------------------------------------------
// Precompute 2: W1A[n][k] = sum_j W1[n][j]*A[j][k]  -> f16   [512x128]
// ---------------------------------------------------------------------------
__global__ void SSM_precompW1A(const float* __restrict__ W1,
                               const float* __restrict__ A,
                               _Float16* __restrict__ W1Ah) {
  int idx = blockIdx.x * blockDim.x + threadIdx.x;   // 65536
  int n = idx >> 7, k = idx & 127;
  float acc = 0.0f;
  for (int j = 0; j < DOUT; ++j) acc += W1[n * DOUT + j] * A[j * DIN + k];
  W1Ah[idx] = (_Float16)acc;
}

// ---------------------------------------------------------------------------
// Precompute 3: convert W2 [128x512] and lin_w [128x128] to f16
// ---------------------------------------------------------------------------
__global__ void SSM_convert(const float* __restrict__ W2,
                            const float* __restrict__ LW,
                            _Float16* __restrict__ W2h,
                            _Float16* __restrict__ LWh) {
  int idx = blockIdx.x * blockDim.x + threadIdx.x;   // 81920
  if (idx < DOUT * DSC) {
    W2h[idx] = (_Float16)W2[idx];
  } else {
    int j = idx - DOUT * DSC;                        // 16384
    LWh[j] = (_Float16)LW[j];
  }
}

// ---------------------------------------------------------------------------
// Main fused kernel: out = relu(u@W1A^T + b1)@W2^T + b2 + u@lin_w^T
// Block: 128 threads (4 wave32), 64 batch rows; each wave owns 16 rows.
// ---------------------------------------------------------------------------
__global__ __launch_bounds__(128)
void SSM_main(const float*    __restrict__ u,
              const _Float16* __restrict__ W1A,   // [512][128]
              const _Float16* __restrict__ W2h,   // [128][512]
              const _Float16* __restrict__ LWh,   // [128][128]
              const float*    __restrict__ b1,
              const float*    __restrict__ b2,
              float*          __restrict__ out) {
  __shared__ __align__(16) _Float16 Uh[MTILE][DIN];    // 16 KB
  __shared__ __align__(16) _Float16 Hh[MTILE][CHUNK];  // 32 KB

  const int  tid  = threadIdx.x;
  const long row0 = (long)blockIdx.x * MTILE;

  // ---- Stage u tile (64x128 f32) -> LDS f16, nontemporal streaming loads --
#pragma unroll
  for (int i = 0; i < 16; ++i) {
    int flat = i * 128 + tid;            // 2048 float4's
    int r  = flat >> 5;                  // 32 float4 per row
    int c4 = flat & 31;
    v4f f = __builtin_nontemporal_load(
        (const v4f*)(u + (row0 + r) * DIN + c4 * 4));
    v4h h = { (_Float16)f[0], (_Float16)f[1], (_Float16)f[2], (_Float16)f[3] };
    *(v4h*)&Uh[r][c4 * 4] = h;
  }
  __syncthreads();

  const int wave = tid >> 5;
  const int lane = tid & 31;
  const int lr   = lane & 15;
  const int hi   = lane >> 4;
  const int kb   = hi * 8;     // A-fragment K sub-offset
  const int nb   = hi * 16;    // B-fragment K sub-offset
  const int m0   = wave * 16;

  const _Float16* Ua = &Uh[m0 + lr][0];

  // ---- GEMM2 accumulators (8 n-tiles of 16), init with b2 ----------------
  v8f acc2[8];
#pragma unroll
  for (int nt = 0; nt < 8; ++nt) {
    float bv = b2[nt * 16 + lr];
#pragma unroll
    for (int j = 0; j < 8; ++j) acc2[nt][j] = bv;
  }

  // ---- Skip path: + u @ lin_w^T ------------------------------------------
#pragma unroll
  for (int nt = 0; nt < 8; ++nt) {
    const _Float16* wr = LWh + (nt * 16 + lr) * DIN;
#pragma unroll
    for (int kt = 0; kt < 4; ++kt) {
      v16h a = load_a(Ua + kt * 32, kb);
      v16h b = load_b(wr + kt * 32, nb);
      acc2[nt] = wmma_f16(a, b, acc2[nt]);
    }
  }

  // ---- DSC in two 256-chunks: GEMM1 (+relu) -> LDS, then GEMM2 partial ---
  for (int c = 0; c < 2; ++c) {
    for (int ntg = 0; ntg < 16; ++ntg) {
      const int n  = c * CHUNK + ntg * 16 + lr;
      const float bv = b1[n];
      v8f acc;
#pragma unroll
      for (int j = 0; j < 8; ++j) acc[j] = bv;
      const _Float16* wr = W1A + n * DIN;
#pragma unroll
      for (int kt = 0; kt < 4; ++kt) {
        v16h a = load_a(Ua + kt * 32, kb);
        v16h b = load_b(wr + kt * 32, nb);
        acc = wmma_f16(a, b, acc);
      }
      // relu + C-layout scatter to LDS (lane's 8 values share column n)
#pragma unroll
      for (int v = 0; v < 8; ++v) {
        float x = acc[v];
        x = x > 0.0f ? x : 0.0f;
        Hh[m0 + v + hi * 8][ntg * 16 + lr] = (_Float16)x;
      }
    }
    // each wave reads back only its own 16 rows -> no barrier needed
    const _Float16* Ha = &Hh[m0 + lr][0];
#pragma unroll
    for (int nt = 0; nt < 8; ++nt) {
      const _Float16* wr = W2h + (nt * 16 + lr) * DSC + c * CHUNK;
#pragma unroll
      for (int kt = 0; kt < 8; ++kt) {
        v16h a = load_a(Ha + kt * 32, kb);
        v16h b = load_b(wr + kt * 32, nb);
        acc2[nt] = wmma_f16(a, b, acc2[nt]);
      }
    }
  }

  // ---- Store out (f32), nontemporal --------------------------------------
#pragma unroll
  for (int nt = 0; nt < 8; ++nt) {
#pragma unroll
    for (int v = 0; v < 8; ++v) {
      long r = row0 + m0 + v + hi * 8;
      __builtin_nontemporal_store(acc2[nt][v], out + r * DOUT + nt * 16 + lr);
    }
  }
}

// ---------------------------------------------------------------------------
// Launch
// ---------------------------------------------------------------------------
extern "C" void kernel_launch(void* const* d_in, const int* in_sizes, int n_in,
                              void* d_out, int out_size, void* d_ws, size_t ws_size,
                              hipStream_t stream) {
  (void)in_sizes; (void)n_in; (void)out_size; (void)ws_size;

  const float* u  = (const float*)d_in[0];
  const float* D  = (const float*)d_in[1];
  // d_in[2] nu_log, d_in[3] theta_log: dead (x0 == 0 => lam term vanishes)
  const float* gl = (const float*)d_in[4];
  const float* Br = (const float*)d_in[5];
  const float* Bi = (const float*)d_in[6];
  const float* Cr = (const float*)d_in[7];
  const float* Ci = (const float*)d_in[8];
  const float* W1 = (const float*)d_in[9];
  const float* b1 = (const float*)d_in[10];
  const float* W2 = (const float*)d_in[11];
  const float* b2 = (const float*)d_in[12];
  const float* LW = (const float*)d_in[13];
  float* out = (float*)d_out;

  char* ws = (char*)d_ws;
  float*    A    = (float*)ws;                                  //  64 KB
  _Float16* W1Ah = (_Float16*)(ws + (64 << 10));                // 128 KB
  _Float16* W2h  = (_Float16*)(ws + (64 << 10) + (128 << 10));  // 128 KB
  _Float16* LWh  = (_Float16*)(ws + (64 << 10) + (256 << 10));  //  32 KB

  SSM_precompA  <<<DOUT * DIN / 256, 256, 0, stream>>>(D, gl, Br, Bi, Cr, Ci, A);
  SSM_precompW1A<<<DSC * DIN  / 256, 256, 0, stream>>>(W1, A, W1Ah);
  SSM_convert   <<<(DOUT * DSC + DOUT * DIN) / 256, 256, 0, stream>>>(W2, LW, W2h, LWh);
  SSM_main      <<<BSZ / MTILE, 128, 0, stream>>>(u, W1Ah, W2h, LWh, b1, b2, out);
}
